// DWAttention_33079838113964
// MI455X (gfx1250) — matrix operand order repaired and checked by reading, not compiled, not run on hardware
//
#include <hip/hip_runtime.h>
#include <hip/hip_bf16.h>
#include <stdint.h>

// ---------------------------------------------------------------------------
// DW-Attention for MI455X (gfx1250): 3 stages, all matmuls on bf16 WMMA with
// fp32 accumulation (v_wmma_f32_16x16x32_bf16), softmax in fp32.
//   Stage 1: fused QKV projection; weight tile staged in LDS (shared by 8
//            waves), X streamed from global with b128+cvt.
//   Stage 2: per-head attention; K/V head tiles staged into LDS once per
//            block via GLOBAL_LOAD_ASYNC_TO_LDS_B128 (ASYNCcnt), softmax in
//            fp32 with intra-half wave32 shuffles, P restaged via LDS.
//   Stage 3: out projection, same structure as stage 1.
// ---------------------------------------------------------------------------

typedef __bf16 v16bf __attribute__((ext_vector_type(16)));
typedef __bf16 v2bf  __attribute__((ext_vector_type(2)));
typedef float  v8f   __attribute__((ext_vector_type(8)));
typedef float  f32x2 __attribute__((ext_vector_type(2)));

#define DEV __device__ __forceinline__

// 16-bit A-matrix (MxK) packed K-pair start for VGPR j, lane half h
// (ISA 7.12.2: lanes0-15 K=0..7,16..23 ; lanes16-31 K=8..15,24..31)
DEV int kpairA(int h, int j) { return ((j < 4) ? (2 * j) : (2 * j + 8)) + 8 * h; }
// 16-bit B-matrix (KxN) packed K-pair start for VGPR j, lane half h
// (same K order as A: lanes0-15 K=0..7,16..23 ; lanes16-31 K=8..15,24..31)
DEV int kpairB(int h, int j) { return ((j < 4) ? (2 * j) : (2 * j + 8)) + 8 * h; }

DEV v8f wmma_bf16(v16bf a, v16bf b, v8f c) {
  return __builtin_amdgcn_wmma_f32_16x16x32_bf16(false, a, false, b, (short)0, c,
                                                 false, false);
}

// Async global->LDS copy of 16 bytes (gfx1250, tracked by ASYNCcnt).
DEV void async_copy_b128(uint32_t lds_off, uint64_t gaddr) {
  asm volatile("global_load_async_to_lds_b128 %0, %1, off"
               :: "v"(lds_off), "v"(gaddr) : "memory");
}
DEV void wait_asynccnt0() {
  asm volatile("s_wait_asynccnt 0x0" ::: "memory");
}
DEV uint32_t lds_addr32(const void* p) {   // LDS aperture: low 32 bits = offset
  return (uint32_t)(uintptr_t)p;
}

// ---------------------------------------------------------------------------
// Stage 1: QKV projection. grid = (M/128, 1152/16), block = 256 (8 waves).
// Weight tile [16 x 384] staged once in LDS (bf16); each wave: one 16x16
// output tile, K-loop of 12 x (K=32) WMMAs.
// ---------------------------------------------------------------------------
__global__ __launch_bounds__(256)
void qkv_proj_kernel(const float* __restrict__ x, const float* __restrict__ W,
                     const float* __restrict__ bias, __bf16* __restrict__ qk,
                     __bf16* __restrict__ vT) {
  __shared__ __bf16 wLds[16 * 384];        // 12 KB [col][k]
  const int tid = threadIdx.x, wave = tid >> 5, lane = tid & 31;
  const int half = lane >> 4, ln = lane & 15;
  const int rowBase = blockIdx.x * 128 + wave * 16;
  const int colBase = blockIdx.y * 16;

  // cooperative stage + convert of the shared weight tile
  for (int i = tid; i < 16 * 384; i += 256)
    wLds[i] = (__bf16)W[(size_t)(colBase + (i / 384)) * 384 + (i % 384)];
  __syncthreads();

  const float* xrow = x + (size_t)(rowBase + ln) * 384;

  v8f acc = {};
#pragma unroll
  for (int kb = 0; kb < 384; kb += 32) {
    if (kb + 32 < 384) __builtin_prefetch(xrow + kb + 32, 0, 0);
    v16bf a, b;
#pragma unroll
    for (int j = 0; j < 8; ++j) {
      f32x2 xa = *(const f32x2*)(xrow + kb + kpairA(half, j));
      v2bf  wb = *(const v2bf*)(wLds + ln * 384 + kb + kpairB(half, j));
      a[2 * j] = (__bf16)xa.x;  a[2 * j + 1] = (__bf16)xa.y;
      b[2 * j] = wb.x;          b[2 * j + 1] = wb.y;
    }
    acc = wmma_bf16(a, b, acc);
  }

  const int o = colBase + ln;     // output channel (D-layout: n = lane%16)
  const float bv = bias[o];
#pragma unroll
  for (int r = 0; r < 8; ++r) {   // D-layout: m = r + 8*half
    const int m = rowBase + r + 8 * half;
    const float val = acc[r] + bv;
    if (o < 768) {                                  // Q / K -> [M x 768]
      qk[(size_t)m * 768 + o] = (__bf16)val;
    } else {                                        // V -> transposed per head
      const int head = (o - 768) >> 5, dim = (o - 768) & 31;
      const int bw = m >> 8, key = m & 255;
      vT[(size_t)((bw * 12 + head) * 32 + dim) * 256 + key] = (__bf16)val;
    }
  }
}

// ---------------------------------------------------------------------------
// Stage 2: attention. grid = 1536 (B*n*H), block = 256 (8 waves).
// K [256x32] and V^T [32x256] staged into LDS once per block with async
// global->LDS b128 copies; each wave owns 16-query row tiles (2/wave).
// ---------------------------------------------------------------------------
__global__ __launch_bounds__(256)
void attn_kernel(const __bf16* __restrict__ qk, const __bf16* __restrict__ vT,
                 __bf16* __restrict__ aout) {
  __shared__ __bf16 kLds[256 * 32];      // 16 KB  [key][dim]
  __shared__ __bf16 vLds[32 * 256];      // 16 KB  [dim][key]
  __shared__ __bf16 pLds[8 * 16 * 32];   //  8 KB  wave-private P chunks
  const int tid = threadIdx.x, wave = tid >> 5, lane = tid & 31;
  const int half = lane >> 4, ln = lane & 15;
  const int bw = blockIdx.x / 12, h = blockIdx.x % 12;
  const __bf16* qkw = qk + (size_t)bw * 256 * 768;
  const __bf16* vhead = vT + (size_t)(bw * 12 + h) * 32 * 256;
  __bf16* ldsW = pLds + wave * (16 * 32);
  const float SCALE = 0.17677669529663687f;   // 1/sqrt(32)

  // ---- async stage: each thread copies one 64B K row + one 64B V segment
  {
    const uint64_t gK = (uint64_t)(uintptr_t)(qkw + (size_t)tid * 768 + 384 + h * 32);
    const uint64_t gV = (uint64_t)(uintptr_t)(vhead + (size_t)tid * 32);
    const uint32_t lK = lds_addr32(kLds + tid * 32);
    const uint32_t lV = lds_addr32(vLds + tid * 32);
#pragma unroll
    for (int c = 0; c < 4; ++c) {
      async_copy_b128(lK + 16u * c, gK + 16u * c);
      async_copy_b128(lV + 16u * c, gV + 16u * c);
    }
    wait_asynccnt0();
  }
  __syncthreads();

  for (int it = 0; it < 2; ++it) {
    const int m0 = (wave + 8 * it) * 16;    // query base within window

    // Q A-fragment (16x32, K = head dim) from global (no cross-wave reuse)
    v16bf qa;
#pragma unroll
    for (int j = 0; j < 8; ++j) {
      const int k = kpairA(half, j);
      v2bf p = *(const v2bf*)(qkw + (size_t)(m0 + ln) * 768 + h * 32 + k);
      qa[2 * j] = p.x;  qa[2 * j + 1] = p.y;
    }

    // S = Q K^T : 16 key tiles of 16, one K=32 WMMA each; K from LDS
    v8f acc[16];
#pragma unroll
    for (int t = 0; t < 16; ++t) {
      v16bf kf;
#pragma unroll
      for (int j = 0; j < 8; ++j) {
        const int k = kpairB(half, j);
        v2bf p = *(const v2bf*)(kLds + (t * 16 + ln) * 32 + k);
        kf[2 * j] = p.x;  kf[2 * j + 1] = p.y;
      }
      v8f z = {};
      acc[t] = wmma_bf16(qa, kf, z);
    }

    // softmax: row m = r + 8*half spans 16 lanes of this half x 16 tiles
    float rinv[8];
#pragma unroll
    for (int r = 0; r < 8; ++r) {
      float m = -3.0e38f;
#pragma unroll
      for (int t = 0; t < 16; ++t) { acc[t][r] *= SCALE; m = fmaxf(m, acc[t][r]); }
#pragma unroll
      for (int off = 1; off < 16; off <<= 1) m = fmaxf(m, __shfl_xor(m, off, 32));
      float s = 0.f;
#pragma unroll
      for (int t = 0; t < 16; ++t) { float p = __expf(acc[t][r] - m); acc[t][r] = p; s += p; }
#pragma unroll
      for (int off = 1; off < 16; off <<= 1) s += __shfl_xor(s, off, 32);
      rinv[r] = 1.0f / s;
    }
#pragma unroll
    for (int t = 0; t < 16; ++t)
#pragma unroll
      for (int r = 0; r < 8; ++r) acc[t][r] *= rinv[r];

    // O = P @ V : K=256 in 8 chunks of 32; P D-layout -> A-layout via LDS
    v8f o0 = {}, o1 = {};
    for (int kk = 0; kk < 8; ++kk) {
#pragma unroll
      for (int t = 0; t < 2; ++t)
#pragma unroll
        for (int r = 0; r < 8; ++r)
          ldsW[(r + 8 * half) * 32 + t * 16 + ln] = (__bf16)acc[2 * kk + t][r];
      __syncthreads();
      v16bf pa;
#pragma unroll
      for (int j = 0; j < 8; ++j) {
        const int k = kpairA(half, j);
        v2bf p = *(const v2bf*)(ldsW + ln * 32 + k);
        pa[2 * j] = p.x;  pa[2 * j + 1] = p.y;
      }
      __syncthreads();
#pragma unroll
      for (int t = 0; t < 2; ++t) {
        v16bf vb;
#pragma unroll
        for (int j = 0; j < 8; ++j) {
          const int k = kk * 32 + kpairB(half, j);   // pairs contiguous in vLds
          v2bf p = *(const v2bf*)(vLds + (t * 16 + ln) * 256 + k);
          vb[2 * j] = p.x;  vb[2 * j + 1] = p.y;
        }
        if (t == 0) o0 = wmma_bf16(pa, vb, o0);
        else        o1 = wmma_bf16(pa, vb, o1);
      }
    }

    // store O (16x32) -> aout bf16 [M x 384]
    const size_t rowG = (size_t)bw * 256 + m0;
#pragma unroll
    for (int r = 0; r < 8; ++r) {
      const size_t m = rowG + r + 8 * half;
      aout[m * 384 + h * 32 + ln]      = (__bf16)o0[r];
      aout[m * 384 + h * 32 + 16 + ln] = (__bf16)o1[r];
    }
  }
}

// ---------------------------------------------------------------------------
// Stage 3: out projection. grid = (M/128, 384/16), block = 256 (8 waves).
// Weight tile [16 x 384] staged once in LDS (bf16).
// ---------------------------------------------------------------------------
__global__ __launch_bounds__(256)
void out_proj_kernel(const __bf16* __restrict__ aout, const float* __restrict__ Wo,
                     float* __restrict__ out) {
  __shared__ __bf16 wLds[16 * 384];        // 12 KB [col][k]
  const int tid = threadIdx.x, wave = tid >> 5, lane = tid & 31;
  const int half = lane >> 4, ln = lane & 15;
  const int rowBase = blockIdx.x * 128 + wave * 16;
  const int colBase = blockIdx.y * 16;

  for (int i = tid; i < 16 * 384; i += 256)
    wLds[i] = (__bf16)Wo[(size_t)(colBase + (i / 384)) * 384 + (i % 384)];
  __syncthreads();

  const __bf16* arow = aout + (size_t)(rowBase + ln) * 384;

  v8f acc = {};
#pragma unroll
  for (int kb = 0; kb < 384; kb += 32) {
    if (kb + 32 < 384) __builtin_prefetch(arow + kb + 32, 0, 0);
    v16bf a, b;
#pragma unroll
    for (int j = 0; j < 8; ++j) {
      v2bf pa = *(const v2bf*)(arow + kb + kpairA(half, j));
      v2bf wb = *(const v2bf*)(wLds + ln * 384 + kb + kpairB(half, j));
      a[2 * j] = pa.x;  a[2 * j + 1] = pa.y;
      b[2 * j] = wb.x;  b[2 * j + 1] = wb.y;
    }
    acc = wmma_bf16(a, b, acc);
  }

#pragma unroll
  for (int r = 0; r < 8; ++r)
    out[(size_t)(rowBase + r + 8 * half) * 384 + colBase + ln] = acc[r];
}

// ---------------------------------------------------------------------------
extern "C" void kernel_launch(void* const* d_in, const int* in_sizes, int n_in,
                              void* d_out, int out_size, void* d_ws, size_t ws_size,
                              hipStream_t stream) {
  (void)in_sizes; (void)n_in; (void)out_size; (void)ws_size;
  const float* x    = (const float*)d_in[0];   // [2,64,256,384]
  const float* Win  = (const float*)d_in[1];   // [1152,384]
  const float* bin  = (const float*)d_in[2];   // [1152]
  const float* Wout = (const float*)d_in[3];   // [384,384]
  float* out = (float*)d_out;

  const size_t M = 2ull * 64 * 256;            // 32768 rows
  char* ws = (char*)d_ws;
  __bf16* qk   = (__bf16*)ws;                                          // M*768
  __bf16* vT   = (__bf16*)(ws + M * 768 * sizeof(__bf16));             // M*384
  __bf16* aout = (__bf16*)(ws + (M * 768 + M * 384) * sizeof(__bf16)); // M*384

  dim3 blk(256);
  qkv_proj_kernel<<<dim3(256, 72), blk, 0, stream>>>(x, Win, bin, qk, vT);
  attn_kernel<<<dim3(1536), blk, 0, stream>>>(qk, vT, aout);
  out_proj_kernel<<<dim3(256, 24), blk, 0, stream>>>(aout, Wout, out);
}
